// Attention_11295763989068
// MI455X (gfx1250) — matrix operand order repaired and checked by reading, not compiled, not run on hardware
//
#include <hip/hip_runtime.h>
#include <hip/hip_bf16.h>

// ---------- constants ----------
#define BATCH 4
#define SEQ   2048
#define NX    1024
#define NHEAD 16
#define HD    64
#define MROWS (BATCH * SEQ)   // 8192
#define N3    (3 * NX)        // 3072

typedef __attribute__((ext_vector_type(16))) _Float16 v16h;
typedef __attribute__((ext_vector_type(8)))  _Float16 v8h;
typedef __attribute__((ext_vector_type(8)))  float    v8f;
typedef __attribute__((ext_vector_type(4)))  int      v4i;

union HF16 { v16h v; v8h h[2]; };

#ifndef __has_builtin
#define __has_builtin(x) 0
#endif
#if __has_builtin(__builtin_amdgcn_global_load_async_to_lds_b128)
#define HAVE_ASYNC_LDS 1
#else
#define HAVE_ASYNC_LDS 0
#endif

#if HAVE_ASYNC_LDS
typedef __attribute__((address_space(1))) v4i gv4i;   // global (device) int32x4
typedef __attribute__((address_space(3))) v4i lv4i;   // LDS int32x4
#endif

// ---- async global->LDS staging (16 bytes per lane per issue) ----
__device__ inline void stage16(const _Float16* g, _Float16* l) {
#if HAVE_ASYNC_LDS
    _Float16* gnc = const_cast<_Float16*>(g);
    __builtin_amdgcn_global_load_async_to_lds_b128(
        (gv4i*)gnc, (lv4i*)l, 0, 0);
#else
    *(v8h*)l = *(const v8h*)g;   // sync fallback: global_load_b128 + ds_store_b128
#endif
}

__device__ inline void wait_async_le16() {   // prev tile done, next (16) may be in flight
#if HAVE_ASYNC_LDS
#if __has_builtin(__builtin_amdgcn_s_wait_asynccnt)
    __builtin_amdgcn_s_wait_asynccnt(16);
#else
    asm volatile("s_wait_asynccnt 16" ::: "memory");
#endif
#endif
}

__device__ inline void wait_async_0() {
#if HAVE_ASYNC_LDS
#if __has_builtin(__builtin_amdgcn_s_wait_asynccnt)
    __builtin_amdgcn_s_wait_asynccnt(0);
#else
    asm volatile("s_wait_asynccnt 0" ::: "memory");
#endif
#endif
}

// WMMA f16 A-operand (16x32, MxK). Lanes 0-15: row=lane, K k0+0..7 / k0+16..23;
// lanes 16-31: row=lane-16, K k0+8..15 / k0+24..31.
__device__ inline v16h load_a_frag(const _Float16* p0, int ld) {
    int lane = threadIdx.x & 31;
    int r    = lane & 15;
    int hf   = (lane >> 4) * 8;
    const _Float16* p = p0 + r * ld + hf;
    HF16 f;
    f.h[0] = *(const v8h*)(p);
    f.h[1] = *(const v8h*)(p + 16);
    return f.v;
}

// WMMA f16 B-operand (32x16, KxN) from the TRANSPOSED operand Bt[N][K].
// lane = column (lane&15); lanes 0-15: K 0..7 / 16..23, lanes 16-31: K 8..15 / 24..31.
__device__ inline v16h load_b_frag(const _Float16* p0, int ld) {
    int lane = threadIdx.x & 31;
    int c    = lane & 15;
    int ks   = (lane >> 4) * 8;
    const _Float16* p = p0 + c * ld + ks;
    HF16 f;
    f.h[0] = *(const v8h*)(p);
    f.h[1] = *(const v8h*)(p + 16);
    return f.v;
}

#define WMMA_F16(a, b, c) \
    __builtin_amdgcn_wmma_f32_16x16x32_f16(false, (a), false, (b), (short)0, (c), false, false)

// ---------- prep kernels ----------
__global__ void cvt_f32_f16_kernel(const float* __restrict__ in,
                                   _Float16* __restrict__ out, int n) {
    int i = blockIdx.x * 256 + threadIdx.x;
    if (i < n) out[i] = (_Float16)in[i];
}

// in: [K][N] row-major f32 ; out: [N][K] row-major f16 (i.e. W^T)
__global__ void transpose_f32_f16_kernel(const float* __restrict__ in,
                                         _Float16* __restrict__ out,
                                         int K, int N) {
    int i = blockIdx.x * 256 + threadIdx.x;
    if (i < K * N) {
        int nn = i / K;
        int kk = i - nn * K;
        out[i] = (_Float16)in[kk * N + nn];
    }
}

// ---------- GEMM helpers: 32x64 tile per wave, double-buffered ----------
__device__ inline void gemm_load_frags(const _Float16* __restrict__ A,
                                       const _Float16* __restrict__ Wt,
                                       int m0, int n0, int k0,
                                       v16h a[2], v16h b[4]) {
    a[0] = load_a_frag(A + m0 * NX + k0, NX);
    a[1] = load_a_frag(A + (m0 + 16) * NX + k0, NX);
#pragma unroll
    for (int t = 0; t < 4; ++t)
        b[t] = load_b_frag(Wt + (n0 + 16 * t) * NX + k0, NX);
}

__device__ inline void gemm_mma(const v16h a[2], const v16h b[4], v8f acc[2][4]) {
#pragma unroll
    for (int mi = 0; mi < 2; ++mi)
#pragma unroll
        for (int t = 0; t < 4; ++t)
            acc[mi][t] = WMMA_F16(a[mi], b[t], acc[mi][t]);
}

// ---------- QKV projection: [8192,1024] x [1024,3072] + bias ----------
__global__ __launch_bounds__(32)
void qkv_gemm_kernel(const _Float16* __restrict__ A,    // [8192][1024]
                     const _Float16* __restrict__ Wt,   // [3072][1024]
                     const float*    __restrict__ bias, // [3072]
                     _Float16* __restrict__ Qb,         // [B][H][S][HD]
                     _Float16* __restrict__ Kb,         // [B][H][S][HD]
                     _Float16* __restrict__ Vtb) {      // [B][H][HD][S]
    const int lane  = threadIdx.x;
    const int nlane = lane & 15;
    const int half  = lane >> 4;
    const int n0 = blockIdx.x * 64;
    const int m0 = blockIdx.y * 32;

    v8f acc[2][4];
#pragma unroll
    for (int t = 0; t < 4; ++t) {
        float bv = bias[n0 + 16 * t + nlane];
#pragma unroll
        for (int mi = 0; mi < 2; ++mi)
#pragma unroll
            for (int j = 0; j < 8; ++j) acc[mi][t][j] = bv;
    }

    v16h a0[2], b0[4], a1[2], b1[4];
    gemm_load_frags(A, Wt, m0, n0, 0, a0, b0);
    for (int k0 = 0; k0 < NX; k0 += 64) {
        gemm_load_frags(A, Wt, m0, n0, k0 + 32, a1, b1);   // prefetch
        gemm_mma(a0, b0, acc);                              // compute current
        if (k0 + 64 < NX) gemm_load_frags(A, Wt, m0, n0, k0 + 64, a0, b0);
        gemm_mma(a1, b1, acc);
    }

#pragma unroll
    for (int t = 0; t < 4; ++t) {
        int n   = n0 + 16 * t + nlane;
        int seg = n >> 10;       // 0=Q 1=K 2=V (uniform across wave)
        int w   = n & 1023;
        int h   = w >> 6;
        int d   = w & 63;
#pragma unroll
        for (int mi = 0; mi < 2; ++mi)
#pragma unroll
            for (int r = 0; r < 8; ++r) {
                int row = m0 + 16 * mi + r + 8 * half;
                int b   = row >> 11;
                int s   = row & 2047;
                _Float16 val = (_Float16)acc[mi][t][r];
                if (seg == 0)
                    Qb[(((b * NHEAD + h) * SEQ) + s) * HD + d] = val;
                else if (seg == 1)
                    Kb[(((b * NHEAD + h) * SEQ) + s) * HD + d] = val;
                else
                    Vtb[((b * NHEAD + h) * HD + d) * SEQ + s] = val;
            }
    }
}

// ---------- flash attention: one wave per (b, h, 32-row query tile) ----------
// K/V tiles are staged into LDS with async global->LDS DMA (ASYNCcnt), double
// buffered: block n+1 streams in while block n is computed.
__global__ __launch_bounds__(32)
void flash_attn_kernel(const _Float16* __restrict__ Qb,
                       const _Float16* __restrict__ Kb,
                       const _Float16* __restrict__ Vtb,
                       _Float16* __restrict__ Aout) {   // [8192][1024] merged heads
    __shared__ __align__(16) _Float16 bufK[2][32 * 64]; // 2 x 4KB key tiles
    __shared__ __align__(16) _Float16 bufV[2][64 * 32]; // 2 x 4KB value tiles
    __shared__ __align__(16) _Float16 ldsP[32 * 32];    // probs staging (C->A layout)

    const int lane  = threadIdx.x;
    const int nlane = lane & 15;
    const int half  = lane >> 4;
    const int q0 = blockIdx.x * 32;
    const int h  = blockIdx.y;
    const int b  = blockIdx.z;

    const _Float16* Qh = Qb  + ((b * NHEAD + h) * SEQ) * HD;
    const _Float16* Kh = Kb  + ((b * NHEAD + h) * SEQ) * HD;
    const _Float16* Vh = Vtb + ((b * NHEAD + h) * HD) * SEQ;

    // V staging geometry: 4 lanes per value row (64B), 8 rows per issue
    const int vrow0 = lane >> 2;        // 0..7
    const int vcol  = (lane & 3) * 8;   // halves

    v16h qa[2][2];
#pragma unroll
    for (int u = 0; u < 2; ++u) {
        qa[u][0] = load_a_frag(Qh + (q0 + 16 * u) * HD + 0, HD);
        qa[u][1] = load_a_frag(Qh + (q0 + 16 * u) * HD + 32, HD);
    }

    v8f accO[2][4];
    float m[2][8], l[2][8];
#pragma unroll
    for (int u = 0; u < 2; ++u) {
#pragma unroll
        for (int t = 0; t < 4; ++t)
#pragma unroll
            for (int j = 0; j < 8; ++j) accO[u][t][j] = 0.0f;
#pragma unroll
        for (int r = 0; r < 8; ++r) { m[u][r] = -3.0e38f; l[u][r] = 0.0f; }
    }

    const int nblk = blockIdx.x + 1;   // keys 0 .. q0+31, blocks of 32

    // stage block 0 into buffer 0: K tile contiguous (4KB), V tile strided
    {
#pragma unroll
        for (int ii = 0; ii < 8; ++ii) {
            int off = ii * 256 + lane * 8;                    // halves
            stage16(Kh + off, &bufK[0][off]);
        }
#pragma unroll
        for (int ii = 0; ii < 8; ++ii) {
            int row = ii * 8 + vrow0;                         // d index
            stage16(Vh + row * SEQ + vcol, &bufV[0][row * 32 + vcol]);
        }
    }

    for (int blk = 0; blk < nblk; ++blk) {
        const int jb  = blk * 32;
        const int cur = blk & 1;
        const int nxt = cur ^ 1;
        const bool more = (blk + 1 < nblk);

        if (more) {   // issue next tile while current is still landing/computing
            const int jn = jb + 32;
#pragma unroll
            for (int ii = 0; ii < 8; ++ii) {
                int off = ii * 256 + lane * 8;
                stage16(Kh + jn * HD + off, &bufK[nxt][off]);
            }
#pragma unroll
            for (int ii = 0; ii < 8; ++ii) {
                int row = ii * 8 + vrow0;
                stage16(Vh + row * SEQ + jn + vcol, &bufV[nxt][row * 32 + vcol]);
            }
            wait_async_le16();   // in-order completion: current tile's 16 are done
        } else {
            wait_async_0();
        }
        __syncthreads();

        // fragments now come from LDS (ds_load_b128)
        v16h kf[4], vf[4];
        kf[0] = load_b_frag(&bufK[cur][0], 64);
        kf[1] = load_b_frag(&bufK[cur][32], 64);
        kf[2] = load_b_frag(&bufK[cur][16 * 64], 64);
        kf[3] = load_b_frag(&bufK[cur][16 * 64 + 32], 64);
#pragma unroll
        for (int t = 0; t < 4; ++t)
            vf[t] = load_b_frag(&bufV[cur][(t * 16) * 32], 32);

#pragma unroll
        for (int u = 0; u < 2; ++u) {
            v8f s0, s1;
#pragma unroll
            for (int j = 0; j < 8; ++j) { s0[j] = 0.0f; s1[j] = 0.0f; }
            s0 = WMMA_F16(qa[u][0], kf[0], s0);
            s0 = WMMA_F16(qa[u][1], kf[1], s0);
            s1 = WMMA_F16(qa[u][0], kf[2], s1);
            s1 = WMMA_F16(qa[u][1], kf[3], s1);

            float alpha[8];
#pragma unroll
            for (int r = 0; r < 8; ++r) {
                int row = q0 + 16 * u + r + 8 * half;
                // scale 1/sqrt(64)=0.125, causal mask -> -10000 (reference semantics)
                float v0 = (jb + nlane <= row)      ? s0[r] * 0.125f : -10000.0f;
                float v1 = (jb + 16 + nlane <= row) ? s1[r] * 0.125f : -10000.0f;
                float bm = fmaxf(v0, v1);
#pragma unroll
                for (int o = 1; o < 16; o <<= 1) bm = fmaxf(bm, __shfl_xor(bm, o, 32));
                float mn = fmaxf(m[u][r], bm);
                float a  = __expf(m[u][r] - mn);
                float p0 = __expf(v0 - mn);
                float p1 = __expf(v1 - mn);
                float rs = p0 + p1;
#pragma unroll
                for (int o = 1; o < 16; o <<= 1) rs += __shfl_xor(rs, o, 32);
                l[u][r] = l[u][r] * a + rs;
                m[u][r] = mn;
                alpha[r] = a;
                int prow = 16 * u + r + 8 * half;
                ldsP[prow * 32 + nlane]      = (_Float16)p0;
                ldsP[prow * 32 + 16 + nlane] = (_Float16)p1;
            }
#pragma unroll
            for (int t = 0; t < 4; ++t)
#pragma unroll
                for (int r = 0; r < 8; ++r) accO[u][t][r] *= alpha[r];
        }

        __syncthreads();
        v16h pa0 = load_a_frag(ldsP, 32);            // C-layout -> A-layout via LDS
        v16h pa1 = load_a_frag(ldsP + 16 * 32, 32);
#pragma unroll
        for (int t = 0; t < 4; ++t) {
            accO[0][t] = WMMA_F16(pa0, vf[t], accO[0][t]);
            accO[1][t] = WMMA_F16(pa1, vf[t], accO[1][t]);
        }
        __syncthreads();
    }

#pragma unroll
    for (int u = 0; u < 2; ++u) {
        float inv[8];
#pragma unroll
        for (int r = 0; r < 8; ++r) inv[r] = 1.0f / l[u][r];
#pragma unroll
        for (int t = 0; t < 4; ++t)
#pragma unroll
            for (int r = 0; r < 8; ++r) {
                int row = b * SEQ + q0 + 16 * u + r + 8 * half;
                int col = h * HD + t * 16 + nlane;
                Aout[row * NX + col] = (_Float16)(accO[u][t][r] * inv[r]);
            }
    }
}

// ---------- output projection: [8192,1024] x [1024,1024] + bias -> f32 ----------
__global__ __launch_bounds__(32)
void proj_gemm_kernel(const _Float16* __restrict__ A,    // [8192][1024]
                      const _Float16* __restrict__ Wt,   // [1024][1024] (W^T)
                      const float*    __restrict__ bias, // [1024]
                      float* __restrict__ out) {         // [8192][1024]
    const int lane  = threadIdx.x;
    const int nlane = lane & 15;
    const int half  = lane >> 4;
    const int n0 = blockIdx.x * 64;
    const int m0 = blockIdx.y * 32;

    v8f acc[2][4];
#pragma unroll
    for (int t = 0; t < 4; ++t) {
        float bv = bias[n0 + 16 * t + nlane];
#pragma unroll
        for (int mi = 0; mi < 2; ++mi)
#pragma unroll
            for (int j = 0; j < 8; ++j) acc[mi][t][j] = bv;
    }

    v16h a0[2], b0[4], a1[2], b1[4];
    gemm_load_frags(A, Wt, m0, n0, 0, a0, b0);
    for (int k0 = 0; k0 < NX; k0 += 64) {
        gemm_load_frags(A, Wt, m0, n0, k0 + 32, a1, b1);
        gemm_mma(a0, b0, acc);
        if (k0 + 64 < NX) gemm_load_frags(A, Wt, m0, n0, k0 + 64, a0, b0);
        gemm_mma(a1, b1, acc);
    }

#pragma unroll
    for (int t = 0; t < 4; ++t)
#pragma unroll
        for (int mi = 0; mi < 2; ++mi)
#pragma unroll
            for (int r = 0; r < 8; ++r) {
                int row = m0 + 16 * mi + r + 8 * half;
                out[row * NX + n0 + 16 * t + nlane] = acc[mi][t][r];
            }
}

// ---------- launch ----------
extern "C" void kernel_launch(void* const* d_in, const int* in_sizes, int n_in,
                              void* d_out, int out_size, void* d_ws, size_t ws_size,
                              hipStream_t stream) {
    const float* hs      = (const float*)d_in[0];  // [4,2048,1024]
    const float* attn_w  = (const float*)d_in[1];  // [1024,3072]
    const float* attn_b  = (const float*)d_in[2];  // [3072]
    const float* proj_w  = (const float*)d_in[3];  // [1024,1024]
    const float* proj_b  = (const float*)d_in[4];  // [1024]
    float* out = (float*)d_out;

    char* ws = (char*)d_ws;
    const size_t MiB = (size_t)1 << 20;
    _Float16* hsf16  = (_Float16*)(ws);               // 16 MiB (reused as Aout)
    _Float16* attnWt = (_Float16*)(ws + 16 * MiB);    //  6 MiB
    _Float16* projWt = (_Float16*)(ws + 22 * MiB);    //  2 MiB
    _Float16* Qb     = (_Float16*)(ws + 24 * MiB);    // 16 MiB
    _Float16* Kb     = (_Float16*)(ws + 40 * MiB);    // 16 MiB
    _Float16* Vtb    = (_Float16*)(ws + 56 * MiB);    // 16 MiB  (total 72 MiB)
    _Float16* Aout   = hsf16;  // safe alias: hsf16 fully consumed before attn writes

    cvt_f32_f16_kernel<<<(MROWS * NX + 255) / 256, 256, 0, stream>>>(hs, hsf16, MROWS * NX);
    transpose_f32_f16_kernel<<<(NX * N3 + 255) / 256, 256, 0, stream>>>(attn_w, attnWt, NX, N3);
    transpose_f32_f16_kernel<<<(NX * NX + 255) / 256, 256, 0, stream>>>(proj_w, projWt, NX, NX);

    qkv_gemm_kernel<<<dim3(N3 / 64, MROWS / 32), 32, 0, stream>>>(
        hsf16, attnWt, attn_b, Qb, Kb, Vtb);

    flash_attn_kernel<<<dim3(SEQ / 32, NHEAD, BATCH), 32, 0, stream>>>(
        Qb, Kb, Vtb, Aout);

    proj_gemm_kernel<<<dim3(NX / 64, MROWS / 32), 32, 0, stream>>>(
        Aout, projWt, proj_b, out);
}
